// Expert_47802986004683
// MI455X (gfx1250) — matrix operand order repaired and checked by reading, not compiled, not run on hardware
//
#include <hip/hip_runtime.h>
#include <cstdint>
#include <cstddef>

// ---------------- problem constants ----------------
constexpr int E  = 8;
constexpr int TOPK = 2;
constexpr int D  = 1024;
constexpr int H  = 2048;
constexpr int T  = 2 * 2048;      // B*S = 4096 tokens
constexpr int TK = T * TOPK;      // 8192 assignments

// ---------------- tiling ----------------
constexpr int BM = 32;            // tokens per block (2 wave-rows of 16)
constexpr int BN = 128;           // output cols per block (4 wave-cols of 32)
constexpr int KSTEP = 32;         // bf16 WMMA K
constexpr int BROW = 40;          // padded LDS row stride in elements (64B data + 16B pad)
constexpr int TILE_E = BN * BROW; // elements per LDS tile buffer

typedef unsigned int        u32;
typedef unsigned short      u16;
typedef unsigned long long  u64;
typedef __attribute__((ext_vector_type(2)))  u32    v2u;
typedef __attribute__((ext_vector_type(4)))  u32    v4u;
typedef __attribute__((ext_vector_type(8)))  u32    v8u;
typedef __attribute__((ext_vector_type(4)))  int    v4i;
typedef __attribute__((ext_vector_type(8)))  int    v8i;
typedef __attribute__((ext_vector_type(4)))  float  v4f;
typedef __attribute__((ext_vector_type(8)))  float  v8f;
typedef __attribute__((ext_vector_type(16))) __bf16 v16bf;

__device__ __forceinline__ u16 f2bf(float f) {
  u32 u = __float_as_uint(f);
  u32 r = u + 0x7FFFu + ((u >> 16) & 1u);   // round-to-nearest-even
  return (u16)(r >> 16);
}

// Build a 16-element bf16 fragment from two 16B-aligned 8-element chunks.
__device__ __forceinline__ v16bf frag16(const u16* p0, const u16* p1) {
  v4u lo = *(const v4u*)p0;
  v4u hi = *(const v4u*)p1;
  v8u w = { lo[0], lo[1], lo[2], lo[3], hi[0], hi[1], hi[2], hi[3] };
  return __builtin_bit_cast(v16bf, w);
}

// ---------------- Tensor Data Mover: 2D bf16 tile -> LDS -------------------
// D# per CDNA5 ISA ch.8: group0 = {count, lds_addr, global_addr, type=2},
// group1 = {data_size=2B + LDS pad(16DW interval, +4DW), tensor dims, tile
// dims, dim0 stride}. Pads each 64B tile row to 80B in LDS (bank-conflict
// avoidance for the ds_load_b128 fragment reads, BROW=40 elements).
// 6-arg builtin form (clang-23 / therock lane): groups 0..3 + extra + cpol.
__device__ __forceinline__ void tdm_load_tile_2d(
    u32 lds_byte_off, const u16* gptr,
    u32 tensor_d0, u32 tensor_d1, u32 stride_elems,
    u32 tile_d0, u32 tile_d1) {
  u64 ga = (u64)(uintptr_t)gptr;
  v4u g0;
  g0[0] = 1u;                                              // count=1, user desc
  g0[1] = lds_byte_off;                                    // lds_addr
  g0[2] = (u32)ga;                                         // global_addr[31:0]
  g0[3] = (u32)((ga >> 32) & 0x1FFFFFFu) | (2u << 30);     // addr[56:32], type=2
  u32 dw0 = (1u << 16)                                     // data_size = 2 bytes
          | (1u << 20)                                     // pad_enable
          | (3u << 22)                                     // pad_interval: 16 DWORDs (64B)
          | (3u << 25);                                    // pad_amount: 4 DWORDs (16B)
  v8i g1;
  g1[0] = (int)dw0;
  g1[1] = (int)((tensor_d0 & 0xFFFFu) << 16);              // dim0[15:0] @ bits[63:48]
  g1[2] = (int)(((tensor_d0 >> 16) & 0xFFFFu) | ((tensor_d1 & 0xFFFFu) << 16));
  g1[3] = (int)(((tensor_d1 >> 16) & 0xFFFFu) | ((tile_d0 & 0xFFFFu) << 16));
  g1[4] = (int)(tile_d1 & 0xFFFFu);                        // tile_dim1 (tile_dim2=0)
  g1[5] = (int)stride_elems;                               // dim0_stride[31:0]
  g1[6] = 0;                                               // dim0_stride[47:32], dim1_stride lo
  g1[7] = 0;
  v4i z4 = { 0, 0, 0, 0 };
  v8i z8 = { 0, 0, 0, 0, 0, 0, 0, 0 };
  __builtin_amdgcn_tensor_load_to_lds(g0, g1, z4, z4, z8, 0);
}

// ---------------- kernel: zero expert counters ----------------
__global__ void zero_cnt_kernel(int* cnt) {
  if (threadIdx.x < E) cnt[threadIdx.x] = 0;
}

// ---------------- kernel: fp32 -> bf16 conversion ----------------
__global__ void cvt_kernel(const float* __restrict__ src, u16* __restrict__ dst, int n4) {
  int i = blockIdx.x * blockDim.x + threadIdx.x;
  if (i >= n4) return;
  v4f v = *(const v4f*)(src + (size_t)i * 4);
  u32 lo = (u32)f2bf(v[0]) | ((u32)f2bf(v[1]) << 16);
  u32 hi = (u32)f2bf(v[2]) | ((u32)f2bf(v[3]) << 16);
  v2u o = { lo, hi };
  *(v2u*)(dst + (size_t)i * 4) = o;
}

// ---------------- kernel: gating (logits, top-2, softmax, routing lists) ---
__global__ void gate_kernel(const float* __restrict__ x,
                            const float* __restrict__ gw,
                            const float* __restrict__ gb,
                            int* __restrict__ cnt,
                            int* __restrict__ lists,
                            float* __restrict__ wgt) {
  int t = blockIdx.x * blockDim.x + threadIdx.x;
  if (t >= T) return;
  float acc[E];
#pragma unroll
  for (int e = 0; e < E; ++e) acc[e] = gb[e];
  const float* xt = x + (size_t)t * D;
  for (int d = 0; d < D; ++d) {
    float xv = xt[d];
#pragma unroll
    for (int e = 0; e < E; ++e) acc[e] += xv * gw[e * D + d];
  }
  int i0 = 0; float v0 = acc[0];
#pragma unroll
  for (int e = 1; e < E; ++e) if (acc[e] > v0) { v0 = acc[e]; i0 = e; }
  int i1 = (i0 == 0) ? 1 : 0; float v1 = acc[i1];
#pragma unroll
  for (int e = 0; e < E; ++e) if (e != i0 && acc[e] > v1) { v1 = acc[e]; i1 = e; }
  float e1  = __expf(v1 - v0);
  float inv = 1.0f / (1.0f + e1);
  int s0 = atomicAdd(&cnt[i0], 1);
  lists[i0 * T + s0] = (t << 1) | 0;
  wgt[t * 2 + 0] = inv;
  int s1 = atomicAdd(&cnt[i1], 1);
  lists[i1 * T + s1] = (t << 1) | 1;
  wgt[t * 2 + 1] = e1 * inv;
}

// ---------------- kernel: GEMM1 fused (h0, h1, SiLU gate) ----------------
// act[aslot, h] = (x@w0[e].T + b0) * silu(x@w1[e].T + b1), bf16 out.
// Weight tiles DMA'd to LDS by the TDM, double-buffered, one tile ahead.
__global__ __launch_bounds__(256) void gemm1_kernel(
    const u16* __restrict__ xb, const u16* __restrict__ w0b,
    const u16* __restrict__ w1b,
    const float* __restrict__ b0, const float* __restrict__ b1,
    const int* __restrict__ cnt, const int* __restrict__ lists,
    u16* __restrict__ actb) {
  const int e  = blockIdx.z;
  const int m0 = blockIdx.y * BM;
  const int Ne = cnt[e];
  if (m0 >= Ne) return;                       // block-uniform
  const int n0 = blockIdx.x * BN;

  struct Smem {
    u16 b[2][2][TILE_E];                      // [buf][matrix][row*BROW + k]
    int ent[BM];
  };
  __shared__ __align__(16) Smem sm;

  if (threadIdx.x < BM) {                     // wave-uniform predicate
    int idx = m0 + threadIdx.x;
    sm.ent[threadIdx.x] = (idx < Ne) ? lists[e * T + idx] : -1;
  }

  const u16* w0base = w0b + ((size_t)e * H + n0) * D;
  const u16* w1base = w1b + ((size_t)e * H + n0) * D;
  const bool issuer = (threadIdx.x < 32);     // wave 0 drives the TDM

  auto issue = [&](int step, int buf) {
    const int kb = step * KSTEP;
    tdm_load_tile_2d((u32)((buf * 2 + 0) * TILE_E * sizeof(u16)),
                     w0base + kb, D, H, D, KSTEP, BN);
    tdm_load_tile_2d((u32)((buf * 2 + 1) * TILE_E * sizeof(u16)),
                     w1base + kb, D, H, D, KSTEP, BN);
  };

  if (issuer) issue(0, 0);
  __syncthreads();                            // ent[] visible

  const int lane = threadIdx.x & 31;
  const int wid  = threadIdx.x >> 5;
  const int wm   = wid >> 2;                  // 0..1  (16-row tile)
  const int wn   = wid & 3;                   // 0..3  (32-col tile)
  const int r    = lane & 15;
  const int hh   = lane >> 4;

  int entA = sm.ent[wm * 16 + r];
  const size_t arow = (size_t)((entA >= 0) ? (entA >> 1) : 0) * D;

  v8f acc00 = {0,0,0,0,0,0,0,0}, acc01 = {0,0,0,0,0,0,0,0};
  v8f acc10 = {0,0,0,0,0,0,0,0}, acc11 = {0,0,0,0,0,0,0,0};

  const int rowA = (wn * 32 + r) * BROW + hh * 16;        // cols j=0 tile
  const int rowB = (wn * 32 + 16 + r) * BROW + hh * 16;   // cols j=1 tile

  auto step = [&](int kb, int buf) {
    // A 16x32 bf16 fragment from global (gathered token rows)
    v16bf a = frag16(xb + arow + kb + hh * 16,
                     xb + arow + kb + hh * 16 + 8);
    // B 32x16 bf16 fragments from LDS (TDM-staged, padded rows)
    const u16* p00 = &sm.b[buf][0][rowA];
    const u16* p01 = &sm.b[buf][0][rowB];
    const u16* p10 = &sm.b[buf][1][rowA];
    const u16* p11 = &sm.b[buf][1][rowB];
    v16bf b00 = frag16(p00, p00 + 8);
    v16bf b01 = frag16(p01, p01 + 8);
    v16bf b10 = frag16(p10, p10 + 8);
    v16bf b11 = frag16(p11, p11 + 8);
    acc00 = __builtin_amdgcn_wmma_f32_16x16x32_bf16(false, a, false, b00, (short)0, acc00, false, false);
    acc01 = __builtin_amdgcn_wmma_f32_16x16x32_bf16(false, a, false, b01, (short)0, acc01, false, false);
    acc10 = __builtin_amdgcn_wmma_f32_16x16x32_bf16(false, a, false, b10, (short)0, acc10, false, false);
    acc11 = __builtin_amdgcn_wmma_f32_16x16x32_bf16(false, a, false, b11, (short)0, acc11, false, false);
  };

  constexpr int NT = D / KSTEP;               // 32 K-tiles
  for (int i = 0; i < NT - 1; ++i) {
    const int buf = i & 1;
    if (issuer) {
      issue(i + 1, buf ^ 1);                  // prefetch next pair
      __builtin_amdgcn_s_wait_tensorcnt((short)2);  // pair i complete
    }
    __syncthreads();                          // tile i ready for all waves
    step(i * KSTEP, buf);
    __syncthreads();                          // all done reading before overwrite
  }
  if (issuer) __builtin_amdgcn_s_wait_tensorcnt((short)0);
  __syncthreads();
  step((NT - 1) * KSTEP, (NT - 1) & 1);

#pragma unroll
  for (int j = 0; j < 2; ++j) {
    const int col = n0 + wn * 32 + j * 16 + r;
    const float bb0 = b0[e * H + col];
    const float bb1 = b1[e * H + col];
    v8f h0a = j ? acc01 : acc00;
    v8f h1a = j ? acc11 : acc10;
#pragma unroll
    for (int i = 0; i < 8; ++i) {
      int ent = sm.ent[wm * 16 + hh * 8 + i]; // C/D layout: M = 8*hh + i
      if (ent >= 0) {
        float h0v = h0a[i] + bb0;
        float h1v = h1a[i] + bb1;
        float sw  = h1v / (1.0f + __expf(-h1v));   // silu
        actb[(size_t)ent * H + col] = f2bf(h0v * sw);
      }
    }
  }
}

// ---------------- kernel: GEMM2 (down-proj, scaled by routing weight) -----
// outbuf[aslot, d] = wgt[aslot] * (act@w2[e].T + b2), fp32 out
__global__ __launch_bounds__(256) void gemm2_kernel(
    const u16* __restrict__ actb, const u16* __restrict__ w2b,
    const float* __restrict__ b2,
    const int* __restrict__ cnt, const int* __restrict__ lists,
    const float* __restrict__ wgt,
    float* __restrict__ outbuf) {
  const int e  = blockIdx.z;
  const int m0 = blockIdx.y * BM;
  const int Ne = cnt[e];
  if (m0 >= Ne) return;
  const int n0 = blockIdx.x * BN;

  struct Smem {
    u16 b[2][TILE_E];                         // [buf][row*BROW + k]
    int ent[BM];
    float w[BM];
  };
  __shared__ __align__(16) Smem sm;

  if (threadIdx.x < BM) {
    int idx = m0 + threadIdx.x;
    int ent = (idx < Ne) ? lists[e * T + idx] : -1;
    sm.ent[threadIdx.x] = ent;
    sm.w[threadIdx.x]   = (ent >= 0) ? wgt[ent] : 0.0f;
  }

  const u16* w2base = w2b + ((size_t)e * D + n0) * H;
  const bool issuer = (threadIdx.x < 32);

  auto issue = [&](int stepi, int buf) {
    tdm_load_tile_2d((u32)(buf * TILE_E * sizeof(u16)),
                     w2base + stepi * KSTEP, H, D, H, KSTEP, BN);
  };

  if (issuer) issue(0, 0);
  __syncthreads();

  const int lane = threadIdx.x & 31;
  const int wid  = threadIdx.x >> 5;
  const int wm   = wid >> 2;
  const int wn   = wid & 3;
  const int r    = lane & 15;
  const int hh   = lane >> 4;

  int entA = sm.ent[wm * 16 + r];
  const size_t arow = (size_t)((entA >= 0) ? entA : 0) * H;

  v8f acc0 = {0,0,0,0,0,0,0,0}, acc1 = {0,0,0,0,0,0,0,0};

  const int rowA = (wn * 32 + r) * BROW + hh * 16;
  const int rowB = (wn * 32 + 16 + r) * BROW + hh * 16;

  auto step = [&](int kb, int buf) {
    v16bf a = frag16(actb + arow + kb + hh * 16,
                     actb + arow + kb + hh * 16 + 8);
    const u16* p0 = &sm.b[buf][rowA];
    const u16* p1 = &sm.b[buf][rowB];
    v16bf bf0 = frag16(p0, p0 + 8);
    v16bf bf1 = frag16(p1, p1 + 8);
    acc0 = __builtin_amdgcn_wmma_f32_16x16x32_bf16(false, a, false, bf0, (short)0, acc0, false, false);
    acc1 = __builtin_amdgcn_wmma_f32_16x16x32_bf16(false, a, false, bf1, (short)0, acc1, false, false);
  };

  constexpr int NT = H / KSTEP;               // 64 K-tiles
  for (int i = 0; i < NT - 1; ++i) {
    const int buf = i & 1;
    if (issuer) {
      issue(i + 1, buf ^ 1);
      __builtin_amdgcn_s_wait_tensorcnt((short)1);
    }
    __syncthreads();
    step(i * KSTEP, buf);
    __syncthreads();
  }
  if (issuer) __builtin_amdgcn_s_wait_tensorcnt((short)0);
  __syncthreads();
  step((NT - 1) * KSTEP, (NT - 1) & 1);

#pragma unroll
  for (int j = 0; j < 2; ++j) {
    const int col = n0 + wn * 32 + j * 16 + r;
    const float bb = b2[e * D + col];
    v8f ac = j ? acc1 : acc0;
#pragma unroll
    for (int i = 0; i < 8; ++i) {
      int mrow = wm * 16 + hh * 8 + i;
      int ent  = sm.ent[mrow];
      if (ent >= 0) {
        outbuf[(size_t)ent * D + col] = (ac[i] + bb) * sm.w[mrow];
      }
    }
  }
}

// ---------------- kernel: combine the two routed expert outputs -----------
__global__ void combine_kernel(const float* __restrict__ buf, float* __restrict__ out) {
  int i = blockIdx.x * blockDim.x + threadIdx.x;     // in float4 units
  size_t base = (size_t)i * 4;
  if (base >= (size_t)T * D) return;
  size_t t = base / D;
  v4f a = *(const v4f*)(buf + (2 * t) * D + (base - t * D));
  v4f b = *(const v4f*)(buf + (2 * t + 1) * D + (base - t * D));
  v4f rv = a + b;
  *(v4f*)(out + base) = rv;
}

// ---------------- launcher ----------------
extern "C" void kernel_launch(void* const* d_in, const int* in_sizes, int n_in,
                              void* d_out, int out_size, void* d_ws, size_t ws_size,
                              hipStream_t stream) {
  const float* x  = (const float*)d_in[0];
  const float* gw = (const float*)d_in[1];
  const float* gb = (const float*)d_in[2];
  const float* w0 = (const float*)d_in[3];
  const float* b0 = (const float*)d_in[4];
  const float* w1 = (const float*)d_in[5];
  const float* b1 = (const float*)d_in[6];
  const float* w2 = (const float*)d_in[7];
  const float* b2 = (const float*)d_in[8];
  float* out = (float*)d_out;

  char* ws = (char*)d_ws;
  u16*   xb     = (u16*)ws;   ws += (size_t)T  * D * 2;
  u16*   w0b    = (u16*)ws;   ws += (size_t)E  * H * D * 2;
  u16*   w1b    = (u16*)ws;   ws += (size_t)E  * H * D * 2;
  u16*   w2b    = (u16*)ws;   ws += (size_t)E  * D * H * 2;
  u16*   actb   = (u16*)ws;   ws += (size_t)TK * H * 2;
  float* outbuf = (float*)ws; ws += (size_t)TK * D * 4;
  int*   cnt    = (int*)ws;   ws += 256;
  int*   lists  = (int*)ws;   ws += (size_t)E * T * 4;
  float* wgt    = (float*)ws; ws += (size_t)TK * 4;

  zero_cnt_kernel<<<1, 32, 0, stream>>>(cnt);

  const int n4x = T * D / 4;
  cvt_kernel<<<(n4x + 255) / 256, 256, 0, stream>>>(x, xb, n4x);
  const int n4w = E * H * D / 4;
  cvt_kernel<<<(n4w + 255) / 256, 256, 0, stream>>>(w0, w0b, n4w);
  cvt_kernel<<<(n4w + 255) / 256, 256, 0, stream>>>(w1, w1b, n4w);
  cvt_kernel<<<(n4w + 255) / 256, 256, 0, stream>>>(w2, w2b, n4w);

  gate_kernel<<<T / 256, 256, 0, stream>>>(x, gw, gb, cnt, lists, wgt);

  gemm1_kernel<<<dim3(H / BN, T / BM, E), 256, 0, stream>>>(
      xb, w0b, w1b, b0, b1, cnt, lists, actb);
  gemm2_kernel<<<dim3(D / BN, T / BM, E), 256, 0, stream>>>(
      actb, w2b, b2, cnt, lists, wgt, outbuf);

  combine_kernel<<<((T * D / 4) + 255) / 256, 256, 0, stream>>>(outbuf, out);
}
